// TSCH_NN_23184233464071
// MI455X (gfx1250) — matrix operand order repaired and checked by reading, not compiled, not run on hardware
//
#include <hip/hip_runtime.h>
#include <hip/hip_bf16.h>
#include <cstdint>

// ---------------------------------------------------------------------------
// WMMA fragment helpers (gfx1250, wave32): D = A(16x32 bf16) x B(32x16 bf16) + C(f32)
// Layouts per CDNA5 ISA 7.12.2.
// ---------------------------------------------------------------------------
typedef __attribute__((ext_vector_type(16))) __bf16 bf16x16;
typedef __attribute__((ext_vector_type(8)))  float  f32x8;

union FragU { bf16x16 v; uint4 q[2]; };

// A 16x32 tile from a row-major tile base (global or LDS), leading dim lda.
// lanes 0-15 hold K = [0..8) u [16..24) of row m=lane; lanes 16-31 hold
// K = [8..16) u [24..32). m clamped for partial tiles (stores are guarded).
__device__ __forceinline__ bf16x16 load_a_frag(const __bf16* tile, int lda, int lane, int rows) {
  int half = lane >> 4;
  int m = lane & 15;
  if (m >= rows) m = rows - 1;
  const __bf16* r = tile + (size_t)m * lda + 8 * half;
  FragU f;
  f.q[0] = *(const uint4*)(r);           // K = 8*half + 0..7
  f.q[1] = *(const uint4*)(r + 16);      // K = 16 + 8*half + 0..7
  return f.v;
}

// B 32x16 tile from transposed weights wt[Ncols][K]. lane col n = lane&15,
// lane holds K = k0 + 8*(lane>>4) + 0..7 and k0 + 16 + 8*(lane>>4) + 0..7.
__device__ __forceinline__ bf16x16 load_b_frag(const __bf16* wt, int K, int col0, int k0, int lane) {
  int half = lane >> 4;
  int n = lane & 15;
  const __bf16* r = wt + (size_t)(col0 + n) * K + k0 + 8 * half;
  FragU f;
  f.q[0] = *(const uint4*)(r);
  f.q[1] = *(const uint4*)(r + 16);
  return f.v;
}

__device__ __forceinline__ f32x8 wmma_bf16(bf16x16 a, bf16x16 b, f32x8 c) {
  return __builtin_amdgcn_wmma_f32_16x16x32_bf16(false, a, false, b, (short)0, c, false, false);
}

// float atomic max via int ordering trick (valid with -inf init, mixed signs)
__device__ __forceinline__ void atomicMaxF(float* addr, float v) {
  if (v >= 0.0f) atomicMax((int*)addr, __float_as_int(v));
  else           atomicMin((unsigned int*)addr, __float_as_uint(v));
}

// ---------------------------------------------------------------------------
// Elementwise / conversion kernels
// ---------------------------------------------------------------------------
__global__ void k_fill(float* p, float v, int n) {
  int i = blockIdx.x * blockDim.x + threadIdx.x;
  if (i < n) p[i] = v;
}

// x [N,20] f32 -> xb [N,32] bf16 zero-padded
__global__ void k_convert_x(const float* __restrict__ x, __bf16* __restrict__ xb, int N) {
  int i = blockIdx.x * blockDim.x + threadIdx.x;
  if (i >= N * 32) return;
  int n = i >> 5, k = i & 31;
  xb[i] = (k < 20) ? (__bf16)x[n * 20 + k] : (__bf16)0.0f;
}

// W [20,512] f32 -> wpad [512][32] bf16 (transposed + K zero-padded to 32)
__global__ void k_convert_wpad(const float* __restrict__ W, __bf16* __restrict__ wpad) {
  int i = blockIdx.x * blockDim.x + threadIdx.x;
  if (i >= 512 * 32) return;
  int c = i >> 5, k = i & 31;
  wpad[i] = (k < 20) ? (__bf16)W[k * 512 + c] : (__bf16)0.0f;
}

// W [K,NC] f32 -> wt [NC][K] bf16 (transpose)
__global__ void k_convert_wt(const float* __restrict__ W, __bf16* __restrict__ wt, int K, int NC) {
  int i = blockIdx.x * blockDim.x + threadIdx.x;
  if (i >= K * NC) return;
  int c = i / K, k = i - c * K;
  wt[i] = (__bf16)W[k * NC + c];
}

// gat [N,512] + bias[512] -> bf16 (reuses h buffer)
__global__ void k_bias_bf(const float* __restrict__ gat, const float* __restrict__ b,
                          __bf16* __restrict__ gbf, int N) {
  int i = blockIdx.x * blockDim.x + threadIdx.x;
  if (i >= N * 512) return;
  gbf[i] = (__bf16)(gat[i] + b[i & 511]);
}

// fused = lrelu(concat(ff0, ff1), 0.1) -> bf16 [N,256]
__global__ void k_fuse(const float* __restrict__ f0, const float* __restrict__ f1,
                       __bf16* __restrict__ fb, int N) {
  int i = blockIdx.x * blockDim.x + threadIdx.x;
  if (i >= N * 256) return;
  int n = i >> 8, c = i & 255;
  float v = (c < 128) ? f0[n * 128 + c] : f1[n * 128 + c - 128];
  v = v > 0.0f ? v : 0.1f * v;
  fb[i] = (__bf16)v;
}

// ---------------------------------------------------------------------------
// WMMA GEMM kernels.  Pattern: cooperative global->LDS stage of the shared
// 16xK A tile (one per block, reused by all waves/col-tiles), fragments built
// from LDS (ds_load_b128), B fragments straight from L2-resident bf16 weights.
// ---------------------------------------------------------------------------

// cooperative copy of rows*lda bf16 (16B granules) into LDS
__device__ __forceinline__ void stage_a_tile(__bf16* sA, const __bf16* gA, int rows, int lda,
                                             int tid, int nthreads) {
  int nq = rows * lda / 8;               // uint4 count (lda multiple of 32)
  const uint4* src = (const uint4*)gA;
  uint4* dst = (uint4*)sA;
  for (int i = tid; i < nq; i += nthreads) dst[i] = src[i];
  __syncthreads();
}

// h = x_bf[N,32] @ Wpad -> h_bf [N,512]  (K=32, one WMMA per tile)
__global__ void k_lin_wmma(const __bf16* __restrict__ xb, const __bf16* __restrict__ wpad,
                           __bf16* __restrict__ hbf, int N) {
  __shared__ __bf16 sA[16 * 32];         // 1 KB
  int lane = threadIdx.x & 31, wave = threadIdx.x >> 5;
  int row0 = blockIdx.x * 16;
  int col0 = (blockIdx.y * 8 + wave) * 16;
  int rows = N - row0; rows = rows > 16 ? 16 : rows;
  stage_a_tile(sA, xb + (size_t)row0 * 32, rows, 32, threadIdx.x, blockDim.x);
  bf16x16 a = load_a_frag(sA, 32, lane, rows);
  bf16x16 b = load_b_frag(wpad, 32, col0, 0, lane);
  f32x8 c = {};
  c = wmma_bf16(a, b, c);
  int half = lane >> 4, n = lane & 15;
  __bf16* o = hbf + (size_t)row0 * 512 + col0 + n;
  if (rows == 16) {                      // uniform fast path: no per-element guards
#pragma unroll
    for (int j = 0; j < 8; ++j) o[(size_t)(j + 8 * half) * 512] = (__bf16)c[j];
  } else {
#pragma unroll
    for (int j = 0; j < 8; ++j) {
      int r = j + 8 * half;
      if (r < rows) o[(size_t)r * 512] = (__bf16)c[j];
    }
  }
}

// ff = gat_bf[N,512] @ ffWt[128][512] + ffb -> f32 [N,128]  (K=512)
__global__ void k_ff_wmma(const __bf16* __restrict__ A, const __bf16* __restrict__ wt,
                          const float* __restrict__ bias, float* __restrict__ out, int N) {
  __shared__ __bf16 sA[16 * 512];        // 16 KB, reused by 8 waves
  int lane = threadIdx.x & 31, wave = threadIdx.x >> 5;
  int row0 = blockIdx.x * 16;
  int col0 = wave * 16;                  // 8 waves cover 128 cols
  int rows = N - row0; rows = rows > 16 ? 16 : rows;
  stage_a_tile(sA, A + (size_t)row0 * 512, rows, 512, threadIdx.x, blockDim.x);
  f32x8 c = {};
#pragma unroll
  for (int k0 = 0; k0 < 512; k0 += 32) {
    bf16x16 a = load_a_frag(sA + k0, 512, lane, rows);
    bf16x16 b = load_b_frag(wt, 512, col0, k0, lane);
    c = wmma_bf16(a, b, c);
  }
  int half = lane >> 4, n = lane & 15;
  float bs = bias[col0 + n];
  float* o = out + (size_t)row0 * 128 + col0 + n;
  if (rows == 16) {
#pragma unroll
    for (int j = 0; j < 8; ++j) o[(size_t)(j + 8 * half) * 128] = c[j] + bs;
  } else {
#pragma unroll
    for (int j = 0; j < 8; ++j) {
      int r = j + 8 * half;
      if (r < rows) o[(size_t)r * 128] = c[j] + bs;
    }
  }
}

// h1 = lrelu(fused_bf[N,256] @ h1Wt[64][256] + b, 0.1) -> f32 [N,64]  (K=256)
__global__ void k_h1_wmma(const __bf16* __restrict__ A, const __bf16* __restrict__ wt,
                          const float* __restrict__ bias, float* __restrict__ out, int N) {
  __shared__ __bf16 sA[16 * 256];        // 8 KB, reused by 4 waves
  int lane = threadIdx.x & 31, wave = threadIdx.x >> 5;
  int row0 = blockIdx.x * 16;
  int col0 = wave * 16;                  // 4 waves cover 64 cols
  int rows = N - row0; rows = rows > 16 ? 16 : rows;
  stage_a_tile(sA, A + (size_t)row0 * 256, rows, 256, threadIdx.x, blockDim.x);
  f32x8 c = {};
#pragma unroll
  for (int k0 = 0; k0 < 256; k0 += 32) {
    bf16x16 a = load_a_frag(sA + k0, 256, lane, rows);
    bf16x16 b = load_b_frag(wt, 256, col0, k0, lane);
    c = wmma_bf16(a, b, c);
  }
  int half = lane >> 4, n = lane & 15;
  float bs = bias[col0 + n];
  float* o = out + (size_t)row0 * 64 + col0 + n;
  if (rows == 16) {
#pragma unroll
    for (int j = 0; j < 8; ++j) {
      float v = c[j] + bs;
      o[(size_t)(j + 8 * half) * 64] = v > 0.0f ? v : 0.1f * v;
    }
  } else {
#pragma unroll
    for (int j = 0; j < 8; ++j) {
      int r = j + 8 * half;
      if (r < rows) {
        float v = c[j] + bs;
        o[(size_t)r * 64] = v > 0.0f ? v : 0.1f * v;
      }
    }
  }
}

// ---------------------------------------------------------------------------
// GAT attention kernels
// ---------------------------------------------------------------------------
// per-(node,head) scores: s_src/s_dst = <h[n,h,:], a_{src,dst}[h,:]>
__global__ void k_scores(const __bf16* __restrict__ hbf, const float* __restrict__ as,
                         const float* __restrict__ ad, float* __restrict__ ssrc,
                         float* __restrict__ sdst, int N) {
  int i = blockIdx.x * blockDim.x + threadIdx.x;
  if (i >= N * 8) return;
  int n = i >> 3, h = i & 7;
  const __bf16* hp = hbf + (size_t)n * 512 + h * 64;
  const float* a1 = as + h * 64;
  const float* a2 = ad + h * 64;
  float s1 = 0.0f, s2 = 0.0f;
#pragma unroll 8
  for (int c = 0; c < 64; ++c) {
    float v = (float)hp[c];
    s1 += v * a1[c];
    s2 += v * a2[c];
  }
  ssrc[i] = s1;
  sdst[i] = s2;
}

// pass 1: segment max of leaky_relu(s_src[src]+s_dst[dst], 0.2) over dst (+self loops)
__global__ void k_edge_max(const int* __restrict__ ei, int E, int N,
                           const float* __restrict__ ssrc, const float* __restrict__ sdst,
                           float* __restrict__ mx) {
  int i = blockIdx.x * blockDim.x + threadIdx.x;
  int tot = E + N;
  if (i >= tot) return;
  int s, d;
  if (i < E) { s = ei[i]; d = ei[E + i]; } else { s = d = i - E; }
#pragma unroll
  for (int h = 0; h < 8; ++h) {
    float e = ssrc[s * 8 + h] + sdst[d * 8 + h];
    e = e > 0.0f ? e : 0.2f * e;
    atomicMaxF(&mx[d * 8 + h], e);
  }
}

// pass 2: denom = segment sum of exp(e - m[dst])
__global__ void k_edge_sum(const int* __restrict__ ei, int E, int N,
                           const float* __restrict__ ssrc, const float* __restrict__ sdst,
                           const float* __restrict__ mx, float* __restrict__ dn) {
  int i = blockIdx.x * blockDim.x + threadIdx.x;
  int tot = E + N;
  if (i >= tot) return;
  int s, d;
  if (i < E) { s = ei[i]; d = ei[E + i]; } else { s = d = i - E; }
#pragma unroll
  for (int h = 0; h < 8; ++h) {
    float e = ssrc[s * 8 + h] + sdst[d * 8 + h];
    e = e > 0.0f ? e : 0.2f * e;
    atomicAdd(&dn[d * 8 + h], __expf(e - mx[d * 8 + h]));
  }
}

// pass 3: gat[dst] += alpha * h[src]; one wave per edge, 16 channels per lane.
// gat (102 MB) is L2-resident on MI455X (192 MB L2), so the f32 atomics
// resolve at L2 rather than HBM.
__global__ void k_edge_aggr(const int* __restrict__ ei, int E, int N,
                            const float* __restrict__ ssrc, const float* __restrict__ sdst,
                            const float* __restrict__ mx, const float* __restrict__ dn,
                            const __bf16* __restrict__ hbf, float* __restrict__ gat) {
  int wave = (int)((blockIdx.x * blockDim.x + threadIdx.x) >> 5);
  int lane = threadIdx.x & 31;
  int tot = E + N;
  if (wave >= tot) return;               // wave-uniform exit (before shfl)
  int s, d;
  if (wave < E) { s = ei[wave]; d = ei[E + wave]; } else { s = d = wave - E; }
  const __bf16* hs = hbf + (size_t)s * 512;
  __builtin_prefetch(hs + lane * 16, 0, 0);   // global_prefetch_b8: pull gathered row lines
  float alpha = 0.0f;
  if (lane < 8) {
    float e = ssrc[s * 8 + lane] + sdst[d * 8 + lane];
    e = e > 0.0f ? e : 0.2f * e;
    alpha = __expf(e - mx[d * 8 + lane]) / (dn[d * 8 + lane] + 1e-16f);
  }
  float* gd = gat + (size_t)d * 512;
#pragma unroll
  for (int i = 0; i < 16; ++i) {
    int c = lane + i * 32;
    float a = __shfl(alpha, c >> 6, 32);     // broadcast per-head alpha
    atomicAdd(&gd[c], a * (float)hs[c]);
  }
}

// ---------------------------------------------------------------------------
// Tail: h2 -> h3 -> graph mean-pool accumulation (atomic), then heads + L2 loss
// ---------------------------------------------------------------------------
__global__ void k_tail(const float* __restrict__ h1o, const int* __restrict__ batch,
                       const float* __restrict__ h2W, const float* __restrict__ h2b,
                       const float* __restrict__ h3W, const float* __restrict__ h3b,
                       float* __restrict__ sums, float* __restrict__ cnts, int N) {
  __shared__ float sW2[64 * 32];
  __shared__ float sW3[32 * 32];
  __shared__ float sb2[32];
  __shared__ float sb3[32];
  for (int i = threadIdx.x; i < 64 * 32; i += blockDim.x) sW2[i] = h2W[i];
  for (int i = threadIdx.x; i < 32 * 32; i += blockDim.x) sW3[i] = h3W[i];
  if (threadIdx.x < 32) { sb2[threadIdx.x] = h2b[threadIdx.x]; sb3[threadIdx.x] = h3b[threadIdx.x]; }
  __syncthreads();
  int n = blockIdx.x * blockDim.x + threadIdx.x;
  if (n >= N) return;
  float v1[64];
#pragma unroll
  for (int k = 0; k < 64; ++k) v1[k] = h1o[(size_t)n * 64 + k];
  float v2[32];
#pragma unroll
  for (int j = 0; j < 32; ++j) {
    float a = sb2[j];
#pragma unroll
    for (int k = 0; k < 64; ++k) a += v1[k] * sW2[k * 32 + j];
    v2[j] = a > 0.0f ? a : 0.1f * a;
  }
  int g = batch[n];
  float* sg = sums + g * 32;
#pragma unroll
  for (int j = 0; j < 32; ++j) {
    float a = sb3[j];
#pragma unroll
    for (int k = 0; k < 32; ++k) a += v2[k] * sW3[k * 32 + j];
    a = a > 0.0f ? a : 0.1f * a;
    atomicAdd(&sg[j], a);
  }
  atomicAdd(&cnts[g], 1.0f);
}

__global__ void k_head(const float* __restrict__ sums, const float* __restrict__ cnts,
                       const float* __restrict__ ceW, const float* __restrict__ ceb,
                       const float* __restrict__ cvW, const float* __restrict__ cvb,
                       float* __restrict__ out) {
  __shared__ float red[64];
  int g = threadIdx.x;     // launched with exactly 64 threads
  float l2 = 0.0f;
  float c = cnts[g];
  c = c > 1.0f ? c : 1.0f;
  float gf[32];
#pragma unroll
  for (int j = 0; j < 32; ++j) {
    gf[j] = sums[g * 32 + j] / c;
    l2 += gf[j] * gf[j];
  }
#pragma unroll
  for (int o = 0; o < 3; ++o) {
    float a = ceb[o], b = cvb[o];
#pragma unroll
    for (int k = 0; k < 32; ++k) {
      a += gf[k] * ceW[k * 3 + o];
      b += gf[k] * cvW[k * 3 + o];
    }
    out[g * 3 + o] = a;             // out_event [64,3]
    out[192 + g * 3 + o] = b;       // out_env   [64,3]
  }
  red[g] = l2;
  __syncthreads();
  for (int s = 32; s > 0; s >>= 1) {
    if (g < s) red[g] += red[g + s];
    __syncthreads();
  }
  if (g == 0) out[384] = red[0] / (64.0f * 32.0f);   // l2_loss
}

// ---------------------------------------------------------------------------
// Host launcher
// ---------------------------------------------------------------------------
static inline int cdiv_ll(long long a, long long b) { return (int)((a + b - 1) / b); }

extern "C" void kernel_launch(void* const* d_in, const int* in_sizes, int n_in,
                              void* d_out, int out_size, void* d_ws, size_t ws_size,
                              hipStream_t stream) {
  const float* x     = (const float*)d_in[0];
  const int*   ei[2] = {(const int*)d_in[1], (const int*)d_in[2]};
  const int*   batch = (const int*)d_in[5];
  const int N    = in_sizes[0] / 20;
  const int Ecnt[2] = {in_sizes[1] / 2, in_sizes[2] / 2};
  const float* W[2]    = {(const float*)d_in[6],  (const float*)d_in[10]};
  const float* asrc[2] = {(const float*)d_in[7],  (const float*)d_in[11]};
  const float* adst[2] = {(const float*)d_in[8],  (const float*)d_in[12]};
  const float* gb[2]   = {(const float*)d_in[9],  (const float*)d_in[13]};
  const float* ffW[2]  = {(const float*)d_in[14], (const float*)d_in[16]};
  const float* ffb[2]  = {(const float*)d_in[15], (const float*)d_in[17]};
  const float* h1W = (const float*)d_in[18]; const float* h1b = (const float*)d_in[19];
  const float* h2W = (const float*)d_in[20]; const float* h2b = (const float*)d_in[21];
  const float* h3W = (const float*)d_in[22]; const float* h3b = (const float*)d_in[23];
  const float* ceW = (const float*)d_in[24]; const float* ceb = (const float*)d_in[25];
  const float* cvW = (const float*)d_in[26]; const float* cvb = (const float*)d_in[27];

  // carve workspace (~255 MB for N=50k)
  uintptr_t p = (uintptr_t)d_ws;
  auto carve = [&](size_t bytes) -> void* {
    void* r = (void*)p;
    p += (bytes + 255) & ~(size_t)255;
    return r;
  };
  __bf16* xb     = (__bf16*)carve((size_t)N * 32 * 2);
  __bf16* wpad   = (__bf16*)carve((size_t)512 * 32 * 2);
  __bf16* ffwt   = (__bf16*)carve((size_t)128 * 512 * 2);
  __bf16* h1wt   = (__bf16*)carve((size_t)64 * 256 * 2);
  __bf16* hbf    = (__bf16*)carve((size_t)N * 512 * 2);   // h, later reused as gat_bf
  float*  gat    = (float*) carve((size_t)N * 512 * 4);
  float*  ssrc   = (float*) carve((size_t)N * 8 * 4);
  float*  sdst   = (float*) carve((size_t)N * 8 * 4);
  float*  mx     = (float*) carve((size_t)N * 8 * 4);
  float*  dn     = (float*) carve((size_t)N * 8 * 4);
  float*  ffo0   = (float*) carve((size_t)N * 128 * 4);
  float*  ffo1   = (float*) carve((size_t)N * 128 * 4);
  float*  ffo[2] = {ffo0, ffo1};
  __bf16* fusedb = (__bf16*)carve((size_t)N * 256 * 2);
  float*  h1o    = (float*) carve((size_t)N * 64 * 4);
  float*  sums   = (float*) carve((size_t)64 * 32 * 4);
  float*  cnts   = (float*) carve((size_t)64 * 4);

  const int mt = (N + 15) / 16;
  const float NEG_INF = -__builtin_inff();

  k_convert_x<<<cdiv_ll((long long)N * 32, 256), 256, 0, stream>>>(x, xb, N);
  k_convert_wt<<<cdiv_ll(256 * 64, 256), 256, 0, stream>>>(h1W, h1wt, 256, 64);
  k_fill<<<cdiv_ll(64 * 32, 256), 256, 0, stream>>>(sums, 0.0f, 64 * 32);
  k_fill<<<1, 64, 0, stream>>>(cnts, 0.0f, 64);

  for (int t = 0; t < 2; ++t) {
    k_convert_wpad<<<cdiv_ll(512 * 32, 256), 256, 0, stream>>>(W[t], wpad);
    k_convert_wt<<<cdiv_ll(512 * 128, 256), 256, 0, stream>>>(ffW[t], ffwt, 512, 128);
    k_fill<<<cdiv_ll((long long)N * 8, 256), 256, 0, stream>>>(mx, NEG_INF, N * 8);
    k_fill<<<cdiv_ll((long long)N * 8, 256), 256, 0, stream>>>(dn, 0.0f, N * 8);
    k_fill<<<cdiv_ll((long long)N * 512, 256), 256, 0, stream>>>(gat, 0.0f, N * 512);

    k_lin_wmma<<<dim3(mt, 4), 256, 0, stream>>>(xb, wpad, hbf, N);
    k_scores<<<cdiv_ll((long long)N * 8, 256), 256, 0, stream>>>(hbf, asrc[t], adst[t], ssrc, sdst, N);

    int tot = Ecnt[t] + N;
    k_edge_max<<<cdiv_ll(tot, 256), 256, 0, stream>>>(ei[t], Ecnt[t], N, ssrc, sdst, mx);
    k_edge_sum<<<cdiv_ll(tot, 256), 256, 0, stream>>>(ei[t], Ecnt[t], N, ssrc, sdst, mx, dn);
    k_edge_aggr<<<cdiv_ll(tot, 8), 256, 0, stream>>>(ei[t], Ecnt[t], N, ssrc, sdst, mx, dn, hbf, gat);

    k_bias_bf<<<cdiv_ll((long long)N * 512, 256), 256, 0, stream>>>(gat, gb[t], hbf, N);
    k_ff_wmma<<<mt, 256, 0, stream>>>(hbf, ffwt, ffb[t], ffo[t], N);
  }

  k_fuse<<<cdiv_ll((long long)N * 256, 256), 256, 0, stream>>>(ffo0, ffo1, fusedb, N);
  k_h1_wmma<<<mt, 128, 0, stream>>>(fusedb, h1wt, h1b, h1o, N);
  k_tail<<<cdiv_ll(N, 256), 256, 0, stream>>>(h1o, batch, h2W, h2b, h3W, h3b, sums, cnts, N);
  k_head<<<1, 64, 0, stream>>>(sums, cnts, ceW, ceb, cvW, cvb, (float*)d_out);
}
